// PartiallyExchangeableNetwork_89232240542161
// MI455X (gfx1250) — matrix operand order repaired and checked by reading, not compiled, not run on hardware
//
#include <hip/hip_runtime.h>

typedef __attribute__((ext_vector_type(16))) __bf16 v16bf;
typedef __attribute__((ext_vector_type(8)))  __bf16 v8bf;
typedef __attribute__((ext_vector_type(8)))  float  v8f;

#define TS    4096
#define LWIN  4094
#define NSAMP 256

#define WMMA_BF16(A, B, C) \
  __builtin_amdgcn_wmma_f32_16x16x32_bf16(false, (A), false, (B), (short)0, (C), false, false)

// Pack f32 accumulator tile half to bf16 (v_cvt_pk_bf16_f32 x4), then ReLU on
// packed bf16 (v_pk_max_num_bf16 x4). ReLU commutes with bf16 round-to-nearest
// and max_num(-0,+0)=+0, so this matches relu-then-convert. Inline asm avoids
// the NaN-canonicalize + scalarized-max patterns the compiler otherwise emits.
static __device__ __forceinline__ v8bf relu_cvt8(v8f c) {
  union { v8bf v; unsigned int u[4]; } t;
#pragma unroll
  for (int i = 0; i < 4; ++i) {
    unsigned int p;
    asm("v_cvt_pk_bf16_f32 %0, %1, %2"
        : "=v"(p) : "v"(c[2 * i]), "v"(c[2 * i + 1]));
    asm("v_pk_max_num_bf16 %0, %1, 0"
        : "=v"(p) : "v"(p));
    t.u[i] = p;
  }
  return t.v;
}

static __device__ __forceinline__ v16bf cat8(v8bf lo, v8bf hi) {
  return __builtin_shufflevector(lo, hi, 0, 1, 2, 3, 4, 5, 6, 7,
                                 8, 9, 10, 11, 12, 13, 14, 15);
}

// ---------------------------------------------------------------------------
// Kernel 1: zero the output_inner accumulator workspace (N x 128 f32)
// ---------------------------------------------------------------------------
__global__ void pen_zero_ws(float* __restrict__ ws, int n) {
  int i = blockIdx.x * blockDim.x + threadIdx.x;
  if (i < n) ws[i] = 0.0f;
}

// ---------------------------------------------------------------------------
// Kernel 2: phi MLP over all windows, summed per sample into ws (atomic f32).
// Transposed formulation: for a 16-window chunk we compute h^T = W^T * X^T
// with v_wmma_f32_16x16x32_bf16. A = W^T tile from LDS, B = activations.
// All LDS operand loads are software-pipelined one step ahead of the WMMA
// that consumes them, so waits stay partial.
// ---------------------------------------------------------------------------
__global__ __launch_bounds__(256) void pen_phi_kernel(
    const float* __restrict__ x,
    const float* __restrict__ w1, const float* __restrict__ b1,
    const float* __restrict__ w2, const float* __restrict__ b2,
    const float* __restrict__ w3, const float* __restrict__ b3,
    float* __restrict__ ws)
{
  __shared__ __align__(32) __bf16 sW1[128 * 32];    // W1^T, K padded 9->32
  __shared__ __align__(32) __bf16 sW2[128 * 128];   // W2^T
  __shared__ __align__(32) __bf16 sW3[128 * 128];   // W3^T
  __shared__ __align__(32) float  sB1[128];
  __shared__ __align__(32) float  sB2[128];
  __shared__ __align__(32) float  sB3[128];
  __shared__ __align__(32) __bf16 sH[8][16 * 128];  // per-wave activation staging

  const int tid = threadIdx.x;

  // ---- stage transposed bf16 weights + biases in LDS ----
  for (int i = tid; i < 128 * 32; i += 256) {
    int col = i >> 5, k = i & 31;
    float v = (k < 9) ? w1[k * 128 + col] : 0.0f;
    sW1[col * 32 + k] = (__bf16)v;
  }
  for (int i = tid; i < 128 * 128; i += 256) {
    int col = i >> 7, k = i & 127;
    sW2[col * 128 + k] = (__bf16)w2[k * 128 + col];
    sW3[col * 128 + k] = (__bf16)w3[k * 128 + col];
  }
  if (tid < 128) { sB1[tid] = b1[tid]; sB2[tid] = b2[tid]; sB3[tid] = b3[tid]; }
  __syncthreads();

  const int lane = tid & 31;
  const int wid  = tid >> 5;
  const int lrow = lane & 15;         // window slot / W^T row within tile
  const int hh   = lane >> 4;         // lane half
  const int gw   = blockIdx.x * 8 + wid;   // 0..4095
  const int n    = gw >> 4;                // sample
  const int j    = gw & 15;                // chunk offset within sample
  if (n >= NSAMP) return;

  const float* xs = x + (size_t)n * (3 * TS);
  __bf16* hst = &sH[wid][0];

  auto LDA1 = [&](int T) -> v16bf {
    const v8bf* a0 = (const v8bf*)&sW1[(T * 16 + lrow) * 32 + hh * 16];
    return cat8(a0[0], a0[1]);
  };
  auto LDA = [&](const __bf16* W, int idx) -> v16bf {
    const int T = idx >> 2, kc = idx & 3;
    const v8bf* a0 = (const v8bf*)&W[(T * 16 + lrow) * 128 + kc * 32 + hh * 16];
    return cat8(a0[0], a0[1]);
  };

  float acc[8][8];
#pragma unroll
  for (int t = 0; t < 8; ++t)
#pragma unroll
    for (int g = 0; g < 8; ++g) acc[t][g] = 0.0f;

  // ---- software-pipelined x loads: chunk s window base = 16*j + 256*s ----
  float xv[9];
  {
    int w0 = 16 * j + lrow;
    if (w0 > LWIN - 1) w0 = LWIN - 1;
#pragma unroll
    for (int f = 0; f < 9; ++f)
      xv[f] = xs[(f % 3) * TS + w0 + (f / 3)];
  }
  bool valid_next = (16 * j + lrow) < LWIN;

  for (int s = 0; s < 16; ++s) {
    const float mval = valid_next ? 1.0f : 0.0f;   // window-validity mask

    // ---- build B1 = X^T chunk (K=9 features padded to 32, N=16 windows) ----
    // B layout: VGPR v, lanes 0-15 hold K=2v,2v+1; lanes 16-31 hold K=16+2v,..
    // Features live at K=0..8 -> only lane-half 0 carries data.
    v16bf B1;
#pragma unroll
    for (int e = 0; e < 16; ++e) B1[e] = (__bf16)0.0f;
    if (hh == 0) {
#pragma unroll
      for (int f = 0; f < 9; ++f) B1[f] = (__bf16)xv[f];
    }

    // ---- prefetch next chunk's x values under the WMMA body ----
    if (s < 15) {
      int wn = 16 * j + 256 * (s + 1) + lrow;
      valid_next = wn < LWIN;
      if (wn > LWIN - 1) wn = LWIN - 1;
#pragma unroll
      for (int f = 0; f < 9; ++f)
        xv[f] = xs[(f % 3) * TS + wn + (f / 3)];
    }

    // ---- layer 1: 8 tiles, operands preloaded one tile ahead ----
    v16bf Bk[4];                       // h1 as next layer's B operands
    {
      v16bf An = LDA1(0);
      v8f c = *(const v8f*)&sB1[hh * 8];
      v8f cN;
#pragma unroll
      for (int T = 0; T < 8; ++T) {
        v16bf A = An;
        if (T < 7) {
          An = LDA1(T + 1);
          cN = *(const v8f*)&sB1[(T + 1) * 16 + hh * 8];
        }
        c = WMMA_BF16(A, B1, c);
        v8bf r = relu_cvt8(c);
        // D layout: lane holds window=lrow, features T*16+8*hh+g contiguous
        *(v8bf*)&hst[lrow * 128 + T * 16 + hh * 8] = r;
        // Bk[q] needs tiles 2q,2q+1 stored -> load right after tile 2q+1
        if (T & 1)
          Bk[T >> 1] = *(const v16bf*)&hst[lrow * 128 + (T >> 1) * 32 + hh * 16];
        if (T < 7) c = cN;
      }
    }

    // ---- layer 2: flattened (T,kc) pipeline ----
    v16bf Bk2[4];                      // h2 as layer-3 B operands
    {
      v16bf An = LDA(sW2, 0);
      v8f c = *(const v8f*)&sB2[hh * 8];
      v8f cN;
#pragma unroll
      for (int idx = 0; idx < 32; ++idx) {
        const int T = idx >> 2, kc = idx & 3;
        v16bf A = An;
        if (idx < 31) An = LDA(sW2, idx + 1);
        if (kc == 0 && T < 7) cN = *(const v8f*)&sB2[(T + 1) * 16 + hh * 8];
        c = WMMA_BF16(A, Bk[kc], c);
        if (kc == 3) {
          v8bf r = relu_cvt8(c);
          *(v8bf*)&hst[lrow * 128 + T * 16 + hh * 8] = r;
          if (T & 1)
            Bk2[T >> 1] =
                *(const v16bf*)&hst[lrow * 128 + (T >> 1) * 32 + hh * 16];
          if (T < 7) c = cN;
        }
      }
    }

    // ---- layer 3 (no relu), masked window-sum into registers ----
    {
      v16bf An = LDA(sW3, 0);
      v8f c = *(const v8f*)&sB3[hh * 8];
      v8f cN;
#pragma unroll
      for (int idx = 0; idx < 32; ++idx) {
        const int T = idx >> 2, kc = idx & 3;
        v16bf A = An;
        if (idx < 31) An = LDA(sW3, idx + 1);
        if (kc == 0 && T < 7) cN = *(const v8f*)&sB3[(T + 1) * 16 + hh * 8];
        c = WMMA_BF16(A, Bk2[kc], c);
        if (kc == 3) {
#pragma unroll
          for (int g = 0; g < 8; ++g)
            acc[T][g] += c[g] * mval;               // contracts to v_fmac_f32
          if (T < 7) c = cN;
        }
      }
    }
  }

  // ---- reduce over the 16 window lanes within each lane-half, then atomics --
#pragma unroll
  for (int T = 0; T < 8; ++T) {
#pragma unroll
    for (int g = 0; g < 8; ++g) {
      float v = acc[T][g];
      v += __shfl_xor(v, 1, 16);
      v += __shfl_xor(v, 2, 16);
      v += __shfl_xor(v, 4, 16);
      v += __shfl_xor(v, 8, 16);
      if (lrow == 0)
        atomicAdd(&ws[n * 128 + T * 16 + hh * 8 + g], v);
    }
  }
}

// ---------------------------------------------------------------------------
// Kernel 3: rho MLP  (N,134) -> relu(@rho_w1+b1) -> @rho_w2+b2 -> (N,10)
// ---------------------------------------------------------------------------
__global__ __launch_bounds__(128) void pen_rho_kernel(
    const float* __restrict__ x, const float* __restrict__ ws,
    const float* __restrict__ rw1, const float* __restrict__ rb1,
    const float* __restrict__ rw2, const float* __restrict__ rb2,
    float* __restrict__ out)
{
  __shared__ float sio[134];
  __shared__ float sg[128];
  const int n = blockIdx.x, tid = threadIdx.x;
  if (tid < 6) sio[tid] = x[(size_t)n * (3 * TS) + (tid >> 1) * TS + (tid & 1)];
  sio[6 + tid] = ws[n * 128 + tid];
  __syncthreads();
  float a = rb1[tid];
  for (int k = 0; k < 134; ++k) a += sio[k] * rw1[k * 128 + tid];
  sg[tid] = fmaxf(a, 0.0f);
  __syncthreads();
  if (tid < 10) {
    float o = rb2[tid];
    for (int jj = 0; jj < 128; ++jj) o += sg[jj] * rw2[jj * 10 + tid];
    out[n * 10 + tid] = o;
  }
}

// ---------------------------------------------------------------------------
extern "C" void kernel_launch(void* const* d_in, const int* in_sizes, int n_in,
                              void* d_out, int out_size, void* d_ws, size_t ws_size,
                              hipStream_t stream) {
  const float* x   = (const float*)d_in[0];
  const float* w1  = (const float*)d_in[1];
  const float* b1  = (const float*)d_in[2];
  const float* w2  = (const float*)d_in[3];
  const float* b2  = (const float*)d_in[4];
  const float* w3  = (const float*)d_in[5];
  const float* b3  = (const float*)d_in[6];
  const float* rw1 = (const float*)d_in[7];
  const float* rb1 = (const float*)d_in[8];
  const float* rw2 = (const float*)d_in[9];
  const float* rb2 = (const float*)d_in[10];
  float* ws  = (float*)d_ws;
  float* out = (float*)d_out;

  pen_zero_ws<<<(NSAMP * 128 + 255) / 256, 256, 0, stream>>>(ws, NSAMP * 128);
  pen_phi_kernel<<<512, 256, 0, stream>>>(x, w1, b1, w2, b2, w3, b3, ws);
  pen_rho_kernel<<<NSAMP, 128, 0, stream>>>(x, ws, rw1, rb1, rw2, rb2, out);
}
